// Gemma3Attention_13039520711510
// MI455X (gfx1250) — matrix operand order repaired and checked by reading, not compiled, not run on hardware
//
#include <hip/hip_runtime.h>

// ---------- types ----------
typedef __attribute__((ext_vector_type(16))) __bf16 v16bf;
typedef __attribute__((ext_vector_type(8)))  __bf16 v8bf;
typedef __attribute__((ext_vector_type(8)))  float  v8f;
typedef unsigned short bfraw;   // bf16 storage (avoid host-side __bf16 scalars)

// ---------- problem constants ----------
constexpr int BATCH = 4, SEQ = 2048, HID = 2560;
constexpr int NH = 8, NKV = 4, HD = 256, WIN = 1024;
constexpr int MS   = BATCH * SEQ;           // 8192 GEMM rows
constexpr int NQKV = NH*HD + 2*NKV*HD;      // 4096 fused qkv cols
constexpr float SOFTCAP = 50.0f;
constexpr float EPSF    = 1e-6f;
constexpr float SCALEF  = 0.0625f;          // 256^-0.5
constexpr float NEGF    = -1e30f;

__device__ __forceinline__ bfraw f2bf(float f) {
  union { float f; unsigned u; } a; a.f = f;
  unsigned r = a.u + 0x7fffu + ((a.u >> 16) & 1u);   // round-to-nearest-even
  return (bfraw)(r >> 16);
}
__device__ __forceinline__ v8bf ld8(const bfraw* p) { return *(const v8bf*)p; }

// async global -> LDS copy of 16 bytes per lane (gfx1250 async-to-LDS path).
__device__ __forceinline__ void async_cp16(unsigned lds, unsigned gofs, const void* base) {
  asm volatile("global_load_async_to_lds_b128 %0, %1, %2"
               :: "v"(lds), "v"(gofs), "s"(base) : "memory");
}
__device__ __forceinline__ void wait_async_le(int n) {
  if (n == 0) asm volatile("s_wait_asynccnt 0x0" ::: "memory");
  else        asm volatile("s_wait_asynccnt 0x8" ::: "memory");
}

// ============================================================
// Conversion / packing kernels
// ============================================================
__global__ void cvt_f32_bf16(const float* __restrict__ in, bfraw* __restrict__ out, int n) {
  int i = blockIdx.x * blockDim.x + threadIdx.x;
  if (i < n) out[i] = f2bf(in[i]);
}

// Wqkv^T bf16 [NQKV][HID] from Wq[HID][2048], Wk[HID][1024], Wv[HID][1024]
__global__ void build_wqkvt(const float* __restrict__ Wq, const float* __restrict__ Wk,
                            const float* __restrict__ Wv, bfraw* __restrict__ out) {
  int idx = blockIdx.x * blockDim.x + threadIdx.x;      // n*HID + k
  if (idx >= NQKV * HID) return;
  int n = idx / HID, k = idx % HID;
  float v;
  if (n < NH*HD)              v = Wq[(size_t)k*(NH*HD) + n];
  else if (n < NH*HD+NKV*HD)  v = Wk[(size_t)k*(NKV*HD) + (n - NH*HD)];
  else                        v = Wv[(size_t)k*(NKV*HD) + (n - NH*HD - NKV*HD)];
  out[idx] = f2bf(v);
}

// Wo^T bf16 [HID][NH*HD] from Wo[NH*HD][HID]
__global__ void build_wot(const float* __restrict__ Wo, bfraw* __restrict__ out) {
  int idx = blockIdx.x * blockDim.x + threadIdx.x;      // n*(NH*HD) + k
  if (idx >= HID * NH*HD) return;
  int n = idx / (NH*HD), k = idx % (NH*HD);
  out[idx] = f2bf(Wo[(size_t)k*HID + n]);
}

// ============================================================
// Generic bf16 WMMA GEMM: C[M][N] = A[M][K] * Bt[N][K]^T  (fp32 out)
// block tile 128x128, BK=32, 8 waves in 4(M) x 2(N), 32x64 per wave.
// Per K-step a wave hoists 2 A-frags + 4 B-frags (12 ds_load_b128)
// and issues 8 WMMAs (1.5 LDS loads per WMMA).
// Double-buffered LDS tiles filled by async global->LDS copies.
// ============================================================
constexpr int GBM = 128, GBN = 128, GBK = 32;

__global__ __launch_bounds__(256)
void gemm_bf16_nt(const bfraw* __restrict__ A, const bfraw* __restrict__ Bt,
                  float* __restrict__ C, int M, int N, int K) {
  __shared__ __align__(16) bfraw As[2][GBM][GBK + 8];
  __shared__ __align__(16) bfraw Bs[2][GBN][GBK + 8];
  const int nbase = blockIdx.x * GBN;
  const int mbase = blockIdx.y * GBM;
  const int t    = threadIdx.x;
  const int wid  = t >> 5, lane = t & 31;
  const int hf   = lane >> 4, l15 = lane & 15;
  const int wm   = wid >> 1, wn = wid & 1;   // wave tile: rows 32*wm.., cols 64*wn..

  const int r0 = t >> 2, r1 = 64 + (t >> 2); // tile rows for this thread's chunks
  const int cc = (t & 3) * 8;                // k-offset of 8-elem chunk

  auto issue = [&](int buf, int k0) {
    async_cp16((unsigned)(size_t)&As[buf][r0][cc],
               (unsigned)(((size_t)(mbase + r0)*K + k0 + cc) * 2), A);
    async_cp16((unsigned)(size_t)&As[buf][r1][cc],
               (unsigned)(((size_t)(mbase + r1)*K + k0 + cc) * 2), A);
    async_cp16((unsigned)(size_t)&Bs[buf][r0][cc],
               (unsigned)(((size_t)(nbase + r0)*K + k0 + cc) * 2), Bt);
    async_cp16((unsigned)(size_t)&Bs[buf][r1][cc],
               (unsigned)(((size_t)(nbase + r1)*K + k0 + cc) * 2), Bt);
  };

  v8f acc[2][4] = {};
  const int nk = K / GBK;
  issue(0, 0);
  for (int ki = 0; ki < nk; ++ki) {
    wait_async_le(0);
    __syncthreads();                       // current buffer visible; prev reads done
    if (ki + 1 < nk) issue((ki + 1) & 1, (ki + 1) * GBK);
    const int buf = ki & 1;

    union { v16bf v; v8bf h2[2]; } av[2], bv[4];
    #pragma unroll
    for (int mi = 0; mi < 2; ++mi) {
      const bfraw* ap = &As[buf][32*wm + 16*mi + l15][0];
      av[mi].h2[0] = ld8(ap + 8*hf);
      av[mi].h2[1] = ld8(ap + 16 + 8*hf);
    }
    #pragma unroll
    for (int ni = 0; ni < 4; ++ni) {
      const bfraw* bp = &Bs[buf][64*wn + 16*ni + l15][0];
      bv[ni].h2[0] = ld8(bp + 8*hf);
      bv[ni].h2[1] = ld8(bp + 16 + 8*hf);
    }
    #pragma unroll
    for (int mi = 0; mi < 2; ++mi)
      #pragma unroll
      for (int ni = 0; ni < 4; ++ni)
        acc[mi][ni] = __builtin_amdgcn_wmma_f32_16x16x32_bf16(
            false, av[mi].v, false, bv[ni].v, (short)0, acc[mi][ni], false, false);
  }
  #pragma unroll
  for (int mi = 0; mi < 2; ++mi)
    #pragma unroll
    for (int ni = 0; ni < 4; ++ni)
      #pragma unroll
      for (int r = 0; r < 8; ++r) {
        int m = mbase + 32*wm + 16*mi + r + 8*hf;
        int n = nbase + 64*wn + 16*ni + l15;
        C[(size_t)m*N + n] = acc[mi][ni][r];
      }
}

// ============================================================
// RMSNorm (q/k heads) + interleaved RoPE + scatter to attention layouts
//   Qb[b][h][s][d] bf16, Kb[b][kh][s][d] bf16, Vt[b][kh][d][s] bf16
// ============================================================
__global__ __launch_bounds__(256)
void norm_rope_scatter(const float* __restrict__ qkv,
                       const float* __restrict__ fcos, const float* __restrict__ fsin,
                       const float* __restrict__ q_scale, const float* __restrict__ k_scale,
                       bfraw* __restrict__ Qb, bfraw* __restrict__ Kb, bfraw* __restrict__ Vt) {
  const int row = blockIdx.x;
  const int b = row / SEQ, s = row % SEQ;
  const int t = threadIdx.x;
  __shared__ float red[128];
  const float* src = qkv + (size_t)row * NQKV;

  for (int h = 0; h < NH + NKV; ++h) {
    const bool isQ = (h < NH);
    const float* hv = src + (isQ ? h*HD : NH*HD + (h - NH)*HD);
    float v0 = 0.0f, v1 = 0.0f;
    if (t < 128) { v0 = hv[2*t]; v1 = hv[2*t+1]; red[t] = v0*v0 + v1*v1; }
    __syncthreads();
    for (int off = 64; off > 0; off >>= 1) {
      if (t < off) red[t] += red[t + off];
      __syncthreads();
    }
    float inv = rsqrtf(red[0] * (1.0f/HD) + EPSF);
    __syncthreads();                                    // all read red[0] before reuse
    if (t < 128) {
      const float* sc = isQ ? q_scale : k_scale;
      float xr = v0 * inv * (1.0f + sc[2*t]);
      float xi = v1 * inv * (1.0f + sc[2*t+1]);
      float c  = fcos[(size_t)s*(HD/2) + t];
      float sn = fsin[(size_t)s*(HD/2) + t];
      bfraw e = f2bf(xr*c - xi*sn);
      bfraw o = f2bf(xr*sn + xi*c);
      size_t base = isQ ? ((size_t)(b*NH  + h)      *SEQ + s)*HD
                        : ((size_t)(b*NKV + (h-NH)) *SEQ + s)*HD;
      bfraw* dst = isQ ? Qb : Kb;
      dst[base + 2*t]     = e;
      dst[base + 2*t + 1] = o;
    }
    __syncthreads();
  }
  #pragma unroll
  for (int kh = 0; kh < NKV; ++kh) {
    float v = src[NH*HD + NKV*HD + kh*HD + t];
    Vt[((size_t)(b*NKV + kh)*HD + t)*SEQ + s] = f2bf(v);
  }
}

// ============================================================
// Flash attention, sliding window + causal + tanh soft-cap
// block: 32 queries of one (b,h); 8 waves; key tiles of 64.
// K/V tiles arrive via async global->LDS; async loads complete in order,
// so waiting ASYNCcnt<=8 releases the K tile while V still streams in
// behind QK + softmax; ASYNCcnt==0 is required only before PV.
// ============================================================
__global__ __launch_bounds__(256)
void attn_kernel(const bfraw* __restrict__ Qb, const bfraw* __restrict__ Kb,
                 const bfraw* __restrict__ Vt, bfraw* __restrict__ Out) {
  constexpr int BQ = 32, TK = 64;
  __shared__ __align__(16) bfraw Qs[BQ][HD + 8];
  __shared__ __align__(16) bfraw Ks[TK][HD + 8];
  __shared__ __align__(16) bfraw Vs[HD][TK + 8];
  __shared__ float Sc[BQ][TK];
  __shared__ __align__(16) bfraw Ps[BQ][TK + 8];
  __shared__ float m_s[BQ], l_s[BQ], alpha_s[BQ];

  const int qbase = blockIdx.x * BQ;
  const int bh = blockIdx.y;
  const int b = bh / NH, h = bh % NH;
  const int kvh = h / (NH / NKV);
  const int t = threadIdx.x;
  const int wid = t >> 5, lane = t & 31;
  const int hf = lane >> 4, l15 = lane & 15;
  const int smi = wid >> 2, sni = wid & 3;    // score tile role
  const int omi = wid & 1,  oni = wid >> 1;   // output tile role

  const size_t qoff = ((size_t)(b*NH + h) * SEQ + qbase) * HD;
  const size_t koff = (size_t)(b*NKV + kvh) * SEQ * HD;
  const size_t voff = (size_t)(b*NKV + kvh) * HD * SEQ;

  #pragma unroll
  for (int c = 0; c < 4; ++c) {               // Q tile: async, 4 chunks/thread
    int ch = t + 256*c;
    int r = ch >> 5, cc = (ch & 31) * 8;
    async_cp16((unsigned)(size_t)&Qs[r][cc],
               (unsigned)((qoff + (size_t)r*HD + cc) * 2), Qb);
  }
  if (t < BQ) { m_s[t] = NEGF; l_s[t] = 0.0f; }

  v8f o_acc[4] = {};

  int klo = qbase - (WIN - 1);
  if (klo < 0) klo = 0;
  klo &= ~(TK - 1);

  for (int kb = klo; kb < qbase + BQ; kb += TK) {
    __syncthreads();                          // prev PV reads of Ks/Vs/Sc done
    #pragma unroll
    for (int c = 0; c < 8; ++c) {             // K tile 64x256 (8 async ops)
      int ch = t + 256*c;
      int r = ch >> 5, cc = (ch & 31) * 8;
      async_cp16((unsigned)(size_t)&Ks[r][cc],
                 (unsigned)((koff + (size_t)(kb + r)*HD + cc) * 2), Kb);
    }
    #pragma unroll
    for (int c = 0; c < 8; ++c) {             // V tile 256x64 (8 async ops)
      int ch = t + 256*c;
      int d = ch >> 3, cc = (ch & 7) * 8;
      async_cp16((unsigned)(size_t)&Vs[d][cc],
                 (unsigned)((voff + (size_t)d*SEQ + kb + cc) * 2), Vt);
    }
    wait_async_le(8);                         // Q + K landed; V still in flight
    __syncthreads();

    // ---- QK^T -> 16x16 score tile per wave ----
    v8f sacc = {};
    #pragma unroll
    for (int dk = 0; dk < HD/32; ++dk) {
      union { v16bf v; v8bf h2[2]; } av, bv;
      const bfraw* ap = &Qs[16*smi + l15][dk*32];
      av.h2[0] = ld8(ap + 8*hf);
      av.h2[1] = ld8(ap + 16 + 8*hf);
      const bfraw* bp = &Ks[16*sni + l15][dk*32];
      bv.h2[0] = ld8(bp + 8*hf);
      bv.h2[1] = ld8(bp + 16 + 8*hf);
      sacc = __builtin_amdgcn_wmma_f32_16x16x32_bf16(
          false, av.v, false, bv.v, (short)0, sacc, false, false);
    }
    #pragma unroll
    for (int r = 0; r < 8; ++r) {             // soft-cap + window mask
      int m  = r + 8*hf;
      int qg = qbase + 16*smi + m;
      int kg = kb + 16*sni + l15;
      float sv = SOFTCAP * tanhf(sacc[r] * (SCALEF / SOFTCAP));
      bool ok = (kg <= qg) && (qg - kg < WIN);
      Sc[16*smi + m][16*sni + l15] = ok ? sv : NEGF;
    }
    __syncthreads();

    // ---- online softmax (threads 0..31, one row each) ----
    if (t < BQ) {
      float mold = m_s[t], mx = mold;
      for (int j = 0; j < TK; ++j) mx = fmaxf(mx, Sc[t][j]);
      float alpha = __expf(mold - mx);
      float rs = 0.0f;
      for (int j = 0; j < TK; ++j) {
        float sv = Sc[t][j];
        float p = (sv > 0.5f*NEGF) ? __expf(sv - mx) : 0.0f;
        Ps[t][j] = f2bf(p);
        rs += p;
      }
      m_s[t] = mx;
      l_s[t] = l_s[t] * alpha + rs;
      alpha_s[t] = alpha;
    }
    wait_async_le(0);                         // V tile fully landed
    __syncthreads();

    // ---- rescale O, then P*V ----
    #pragma unroll
    for (int j = 0; j < 4; ++j)
      #pragma unroll
      for (int r = 0; r < 8; ++r)
        o_acc[j][r] *= alpha_s[16*omi + r + 8*hf];

    #pragma unroll
    for (int ks = 0; ks < 2; ++ks) {
      union { v16bf v; v8bf h2[2]; } av;
      const bfraw* ap = &Ps[16*omi + l15][ks*32];
      av.h2[0] = ld8(ap + 8*hf);
      av.h2[1] = ld8(ap + 16 + 8*hf);
      #pragma unroll
      for (int j = 0; j < 4; ++j) {
        union { v16bf v; v8bf h2[2]; } bv;
        const bfraw* bp = &Vs[64*oni + 16*j + l15][ks*32];
        bv.h2[0] = ld8(bp + 8*hf);
        bv.h2[1] = ld8(bp + 16 + 8*hf);
        o_acc[j] = __builtin_amdgcn_wmma_f32_16x16x32_bf16(
            false, av.v, false, bv.v, (short)0, o_acc[j], false, false);
      }
    }
  }
  __syncthreads();

  #pragma unroll
  for (int j = 0; j < 4; ++j)
    #pragma unroll
    for (int r = 0; r < 8; ++r) {
      int m = r + 8*hf;
      int q = qbase + 16*omi + m;
      float l = l_s[16*omi + m];
      float o = o_acc[j][r] * (l > 0.0f ? 1.0f/l : 0.0f);
      int dcol = 64*oni + 16*j + l15;
      Out[((size_t)(b*SEQ + q))*(NH*HD) + h*HD + dcol] = f2bf(o);
    }
}

// ============================================================
// host launcher
// ============================================================
extern "C" void kernel_launch(void* const* d_in, const int* in_sizes, int n_in,
                              void* d_out, int out_size, void* d_ws, size_t ws_size,
                              hipStream_t stream) {
  (void)in_sizes; (void)n_in; (void)out_size; (void)ws_size;
  const float* x       = (const float*)d_in[0];
  const float* fcos    = (const float*)d_in[1];
  const float* fsin    = (const float*)d_in[2];
  const float* Wq      = (const float*)d_in[3];
  const float* Wk      = (const float*)d_in[4];
  const float* Wv      = (const float*)d_in[5];
  const float* Wo      = (const float*)d_in[6];
  const float* q_scale = (const float*)d_in[7];
  const float* k_scale = (const float*)d_in[8];

  char* ws = (char*)d_ws;
  size_t off = 0;
  auto alloc = [&](size_t bytes) -> void* {
    void* p = ws + off;
    off += (bytes + 255) & ~(size_t)255;
    return p;
  };
  bfraw* xb    = (bfraw*)alloc((size_t)MS * HID * 2);
  bfraw* Wqkvt = (bfraw*)alloc((size_t)NQKV * HID * 2);
  bfraw* Wot   = (bfraw*)alloc((size_t)HID * (NH*HD) * 2);
  float* qkv   = (float*)alloc((size_t)MS * NQKV * 4);
  bfraw* Qb    = (bfraw*)alloc((size_t)BATCH * NH  * SEQ * HD * 2);
  bfraw* Kb    = (bfraw*)alloc((size_t)BATCH * NKV * SEQ * HD * 2);
  bfraw* Vt    = (bfraw*)alloc((size_t)BATCH * NKV * HD * SEQ * 2);
  bfraw* attn  = (bfraw*)alloc((size_t)MS * (NH*HD) * 2);

  {
    int n = MS * HID;
    cvt_f32_bf16<<<(n + 255)/256, 256, 0, stream>>>(x, xb, n);
  }
  build_wqkvt<<<(NQKV*HID + 255)/256, 256, 0, stream>>>(Wq, Wk, Wv, Wqkvt);
  build_wot  <<<(HID*NH*HD + 255)/256, 256, 0, stream>>>(Wo, Wot);

  // fused QKV projection: [8192 x 2560] x [2560 x 4096]
  gemm_bf16_nt<<<dim3(NQKV/GBN, MS/GBM), 256, 0, stream>>>(xb, Wqkvt, qkv, MS, NQKV, HID);

  // RMSNorm + RoPE + scatter to attention layouts
  norm_rope_scatter<<<MS, 256, 0, stream>>>(qkv, fcos, fsin, q_scale, k_scale, Qb, Kb, Vt);

  // sliding-window flash attention
  attn_kernel<<<dim3(SEQ/32, BATCH*NH), 256, 0, stream>>>(Qb, Kb, Vt, attn);

  // output projection: [8192 x 2048] x [2048 x 2560] -> fp32 out
  gemm_bf16_nt<<<dim3(HID/GBN, MS/GBM), 256, 0, stream>>>(attn, Wot, (float*)d_out, MS, HID, NH*HD);
}
